// BertSelfAttentionSubstitute_64673617543197
// MI455X (gfx1250) — matrix operand order repaired and checked by reading, not compiled, not run on hardware
//
#include <hip/hip_runtime.h>
#include <hip/hip_bf16.h>

// ---------------------------------------------------------------------------
// BERT self-attention substitute (relu^2 "softmax") for gfx1250 / MI455X.
// All three GEMMs (QKV projection, Q@K^T, P@V) run on v_wmma_f32_16x16x32_bf16.
// Attention stages K/V tiles into LDS with async global->LDS loads (ASYNCcnt),
// double-buffered, shared by 4 waves working on 64 queries of one (b,h).
// B=4, S=2048, H=16, hd=64, D=1024.
// ---------------------------------------------------------------------------

#define BATCH   4
#define SEQ     2048
#define HEADS   16
#define HDIM    64
#define DMODEL  1024
#define MTOK    (BATCH * SEQ)          // 8192 token rows

#define AS1 __attribute__((address_space(1)))
#define AS3 __attribute__((address_space(3)))

#if defined(__has_builtin)
#  if __has_builtin(__builtin_amdgcn_global_load_async_to_lds_b128)
#    define USE_ASYNC_LDS 1
#  else
#    define USE_ASYNC_LDS 0
#  endif
#else
#  define USE_ASYNC_LDS 0
#endif

typedef __attribute__((ext_vector_type(16))) __bf16 bf16x16;
typedef __attribute__((ext_vector_type(8)))  __bf16 bf16x8;
typedef __attribute__((ext_vector_type(8)))  float  v8f;
typedef __attribute__((ext_vector_type(4)))  int    i32x4;

union Frag {
    bf16x16 f;
    bf16x8  h[2];
};

// float -> bf16 bits with round-to-nearest-even
__device__ __forceinline__ unsigned short f2bf(float x) {
    unsigned u = __float_as_uint(x);
    return (unsigned short)((u + 0x7fffu + ((u >> 16) & 1u)) >> 16);
}

// --- A-fragment: 16x32 bf16, M x K, rows of memory = rows of A ---------------
// ISA layout: lanes 0-15 hold M=0..15, VGPRs 0-3 = K 0..7, VGPRs 4-7 = K 16..23;
//             lanes 16-31 same M, K 8..15 and 24..31.
__device__ __forceinline__ bf16x16 load_a_frag(const unsigned short* base, int ld) {
    const int lane = threadIdx.x & 31;
    const int m    = lane & 15;
    const int half = lane >> 4;
    const unsigned short* p = base + m * ld + half * 8;
    Frag f;
    f.h[0] = *(const bf16x8*)(p);
    f.h[1] = *(const bf16x8*)(p + 16);
    return f.f;
}

// --- B-fragment: 32x16 bf16, K x N, where column n of B is row n of memory ---
// ISA layout: lanes 0-15 hold N=0..15, VGPRs 0-3 = K 0..7, VGPRs 4-7 = K 16..23;
//             lanes 16-31 same N, K 8..15 and 24..31.
__device__ __forceinline__ bf16x16 load_bT_frag(const unsigned short* base, int ld) {
    const int lane = threadIdx.x & 31;
    const int n    = lane & 15;
    const int half = lane >> 4;
    const unsigned short* p = base + n * ld + half * 8;
    Frag f;
    f.h[0] = *(const bf16x8*)(p);
    f.h[1] = *(const bf16x8*)(p + 16);
    return f.f;
}

__device__ __forceinline__ v8f wmma_bf16(bf16x16 a, bf16x16 b, v8f c) {
    return __builtin_amdgcn_wmma_f32_16x16x32_bf16(
        /*neg_a=*/false, a, /*neg_b=*/false, b,
        /*c_mod=*/(short)0, c, /*reuse_a=*/false, /*reuse_b=*/false);
}

// 16-byte global -> LDS move, async (ASYNCcnt) when available
__device__ __forceinline__ void g2l_16B(const unsigned short* src, unsigned short* dst) {
#if USE_ASYNC_LDS
    __builtin_amdgcn_global_load_async_to_lds_b128((AS1 i32x4*)src, (AS3 i32x4*)dst, 0, 0);
#else
    *(uint4*)dst = *(const uint4*)src;
#endif
}

__device__ __forceinline__ void wait_async_stage() {
#if USE_ASYNC_LDS
#  if __has_builtin(__builtin_amdgcn_s_wait_asynccnt)
    __builtin_amdgcn_s_wait_asynccnt(0);
#  else
    asm volatile("s_wait_asynccnt 0" ::: "memory");
#  endif
#endif
}

// ---------------------------------------------------------------------------
// Kernel 1: f32 -> bf16 conversion (4 elements / thread)
// ---------------------------------------------------------------------------
__global__ void cvt_f32_bf16(const float* __restrict__ src,
                             unsigned short* __restrict__ dst, int n) {
    int i = (blockIdx.x * blockDim.x + threadIdx.x) * 4;
    if (i + 3 < n) {
        float4 v = *(const float4*)(src + i);
        dst[i + 0] = f2bf(v.x);
        dst[i + 1] = f2bf(v.y);
        dst[i + 2] = f2bf(v.z);
        dst[i + 3] = f2bf(v.w);
    }
}

// ---------------------------------------------------------------------------
// Kernel 2: fused QKV projection. Y = X @ W^T + b.
//   blockIdx.z selects Q / K / V. One wave computes a 16x64 output tile.
//   Q,K written row-major [B,S,D] bf16. V written transposed [B,H,64,S] bf16
//   so the attention PV GEMM can use the contiguous transposed-B loader.
// ---------------------------------------------------------------------------
__global__ void __launch_bounds__(128)
qkv_proj(const unsigned short* __restrict__ X,
         const unsigned short* __restrict__ Wq, const float* __restrict__ bq, unsigned short* __restrict__ Qo,
         const unsigned short* __restrict__ Wk, const float* __restrict__ bk, unsigned short* __restrict__ Ko,
         const unsigned short* __restrict__ Wv, const float* __restrict__ bv, unsigned short* __restrict__ Vt) {
    const int wave = threadIdx.x >> 5;
    const int lane = threadIdx.x & 31;
    const int n16  = lane & 15;
    const int half = lane >> 4;

    const unsigned short* W;
    const float* bias;
    unsigned short* out;
    bool vtrans = false;
    if (blockIdx.z == 0)      { W = Wq; bias = bq; out = Qo; }
    else if (blockIdx.z == 1) { W = Wk; bias = bk; out = Ko; }
    else                      { W = Wv; bias = bv; out = Vt; vtrans = true; }

    const int tile = blockIdx.x * 4 + wave;       // 8192 tiles: 512 m-tiles x 16 n64-tiles
    const int nt   = tile & 15;
    const int mt   = tile >> 4;
    const int m0   = mt * 16;
    const int n0   = nt * 64;

    v8f acc[4] = {v8f{}, v8f{}, v8f{}, v8f{}};

    for (int kk = 0; kk < DMODEL; kk += 32) {
        bf16x16 a = load_a_frag(X + m0 * DMODEL + kk, DMODEL);
#pragma unroll
        for (int j = 0; j < 4; ++j) {
            bf16x16 b = load_bT_frag(W + (n0 + j * 16) * DMODEL + kk, DMODEL);
            acc[j] = wmma_bf16(a, b, acc[j]);
        }
    }

#pragma unroll
    for (int j = 0; j < 4; ++j) {
        const int col = n0 + j * 16 + n16;
        const float bs = bias[col];
#pragma unroll
        for (int r = 0; r < 8; ++r) {
            const int row = m0 + r + half * 8;      // token index in [0, B*S)
            const unsigned short bf = f2bf(acc[j][r] + bs);
            if (!vtrans) {
                out[row * DMODEL + col] = bf;
            } else {
                const int h = col >> 6;             // head
                const int d = col & 63;             // dim within head
                const int bb = row >> 11;           // token / SEQ
                const int s  = row & 2047;          // token % SEQ
                out[(((bb * HEADS + h) * HDIM) + d) * SEQ + s] = bf;
            }
        }
    }
}

// ---------------------------------------------------------------------------
// Kernel 3: streaming attention.
//   Block = 4 waves = 64 queries of one (b,h). Per 32-key step, the block
//   stages K tile (32x64 bf16) + V tile (64x32 bf16) into double-buffered LDS
//   with async global->LDS loads; every wave then runs 4 WMMAs for S = Q@K^T,
//   relu^2 in-register, a wave-private LDS re-layout of P, and 4 WMMAs for
//   ctx += P@V, all off LDS operands.
// ---------------------------------------------------------------------------
__global__ void __launch_bounds__(128)
attn_relusq(const unsigned short* __restrict__ Q,
            const unsigned short* __restrict__ K,
            const unsigned short* __restrict__ Vt,
            const float* __restrict__ mask,
            float* __restrict__ out) {
    __shared__ __align__(16) unsigned short bufK[2][32 * 64];   // 4 KB each
    __shared__ __align__(16) unsigned short bufV[2][64 * 32];   // 4 KB each
    __shared__ __align__(16) unsigned short plds[4][16 * 32];   // per-wave P tile

    const int tid  = threadIdx.x;
    const int wave = tid >> 5;
    const int lane = tid & 31;
    const int n16  = lane & 15;
    const int half = lane >> 4;

    const int blk = blockIdx.x;          // 2048 = B * H * (S/64)
    const int qb  = blk & 31;            // 32 query-blocks of 64 rows
    const int h   = (blk >> 5) & 15;
    const int b   = blk >> 9;
    const int q0  = qb * 64 + wave * 16;

    const float scale = 0.125f;          // 1/sqrt(64)

    const unsigned short* Kbase0 = K  + (size_t)(b * SEQ) * DMODEL + h * HDIM;
    const unsigned short* Vbase0 = Vt + (size_t)(b * HEADS + h) * HDIM * SEQ;

    const unsigned short* qbase = Q + ((size_t)(b * SEQ + q0) * DMODEL + h * HDIM);
    const bf16x16 aq0 = load_a_frag(qbase, DMODEL);
    const bf16x16 aq1 = load_a_frag(qbase + 32, DMODEL);

    v8f ctx[4] = {v8f{}, v8f{}, v8f{}, v8f{}};
    unsigned short* pw = plds[wave];

    // ---- cooperative stage of one 32-key K tile + V tile into LDS ----------
    // K tile: 32 rows x 64 dims  -> bufK (ld 64);  8 x 16B chunks per row.
    // V tile: 64 rows x 32 keys  -> bufV (ld 32);  4 x 16B chunks per row.
    auto stage = [&](int k0, int buf) {
#pragma unroll
        for (int base = 0; base < 256; base += 128) {
            const int c  = base + tid;
            const int kr = c >> 3, ko = (c & 7) * 8;
            g2l_16B(Kbase0 + (size_t)(k0 + kr) * DMODEL + ko,
                    &bufK[buf][kr * 64 + ko]);
            const int vr = c >> 2, vo = (c & 3) * 8;
            g2l_16B(Vbase0 + (size_t)vr * SEQ + k0 + vo,
                    &bufV[buf][vr * 32 + vo]);
        }
    };

    stage(0, 0);
    wait_async_stage();
    __syncthreads();

    for (int it = 0; it < SEQ / 32; ++it) {
        const int cur = it & 1;
        if (it + 1 < SEQ / 32) stage((it + 1) * 32, cur ^ 1);   // overlap next stage

        const unsigned short* kl = bufK[cur];
        const unsigned short* vl = bufV[cur];
        const int k0 = it * 32;

        // ---- scores for two 16x16 key sub-tiles, relu^2, park P in LDS ----
#pragma unroll
        for (int t = 0; t < 2; ++t) {
            v8f s = v8f{};
            s = wmma_bf16(aq0, load_bT_frag(kl + t * 16 * 64, 64), s);
            s = wmma_bf16(aq1, load_bT_frag(kl + t * 16 * 64 + 32, 64), s);
            const float mval = mask[b * SEQ + k0 + t * 16 + n16];
#pragma unroll
            for (int r = 0; r < 8; ++r) {
                float v = s[r] * scale + mval;
                v = fmaxf(v, 0.0f);
                v = v * v;
                pw[(r + half * 8) * 32 + t * 16 + n16] = f2bf(v);
            }
        }
        // ---- reload P as 16x32 A-fragment (per-wave DS ops are in-order) --
        Frag pa;
        {
            const unsigned short* p = pw + n16 * 32 + half * 8;
            pa.h[0] = *(const bf16x8*)(p);
            pa.h[1] = *(const bf16x8*)(p + 16);
        }
        // ---- ctx += P @ V ------------------------------------------------
#pragma unroll
        for (int j = 0; j < 4; ++j) {
            ctx[j] = wmma_bf16(pa.f, load_bT_frag(vl + j * 16 * 32, 32), ctx[j]);
        }

        wait_async_stage();     // next tile fully in LDS
        __syncthreads();        // everyone done with 'cur' before it is reused
    }

    // ---- write ctx: out[b, q, h*64 + d] (f32) ------------------------------
#pragma unroll
    for (int j = 0; j < 4; ++j) {
#pragma unroll
        for (int r = 0; r < 8; ++r) {
            const int row = b * SEQ + q0 + r + half * 8;
            out[(size_t)row * DMODEL + h * HDIM + j * 16 + n16] = ctx[j][r];
        }
    }
}

// ---------------------------------------------------------------------------
// Host launch
// ---------------------------------------------------------------------------
extern "C" void kernel_launch(void* const* d_in, const int* in_sizes, int n_in,
                              void* d_out, int out_size, void* d_ws, size_t ws_size,
                              hipStream_t stream) {
    const float* hs   = (const float*)d_in[0];   // [4,2048,1024]
    const float* mask = (const float*)d_in[1];   // [4,1,1,2048]
    const float* Wq   = (const float*)d_in[2];
    const float* bq   = (const float*)d_in[3];
    const float* Wk   = (const float*)d_in[4];
    const float* bk   = (const float*)d_in[5];
    const float* Wv   = (const float*)d_in[6];
    const float* bv   = (const float*)d_in[7];
    float* out        = (float*)d_out;

    const int nHS = MTOK * DMODEL;               // 8,388,608
    const int nW  = DMODEL * DMODEL;             // 1,048,576

    unsigned short* hsb = (unsigned short*)d_ws;
    unsigned short* wqb = hsb + nHS;
    unsigned short* wkb = wqb + nW;
    unsigned short* wvb = wkb + nW;
    unsigned short* qb  = wvb + nW;
    unsigned short* kb  = qb + nHS;
    unsigned short* vtb = kb + nHS;              // total ~73.4 MB of workspace

    // 1) f32 -> bf16 conversions
    cvt_f32_bf16<<<nHS / (256 * 4), 256, 0, stream>>>(hs, hsb, nHS);
    cvt_f32_bf16<<<nW  / (256 * 4), 256, 0, stream>>>(Wq, wqb, nW);
    cvt_f32_bf16<<<nW  / (256 * 4), 256, 0, stream>>>(Wk, wkb, nW);
    cvt_f32_bf16<<<nW  / (256 * 4), 256, 0, stream>>>(Wv, wvb, nW);

    // 2) fused QKV projection: 8192 tiles per projection, 4 waves/block, z = Q/K/V
    {
        dim3 grid(MTOK / 16 * (DMODEL / 64) / 4, 1, 3);   // (2048, 1, 3)
        qkv_proj<<<grid, 128, 0, stream>>>(hsb,
                                           wqb, bq, qb,
                                           wkb, bk, kb,
                                           wvb, bv, vtb);
    }

    // 3) attention: B*H*(S/64) = 2048 blocks, 4 waves each
    attn_relusq<<<BATCH * HEADS * (SEQ / 64), 128, 0, stream>>>(
        qb, kb, vtb, mask, out);
}
